// MultitaskLearner_72988674228458
// MI455X (gfx1250) — compile-verified, matrix operand order checked by reading, not run on hardware
//
#include <hip/hip_runtime.h>
#include <hip/hip_bf16.h>

typedef __attribute__((ext_vector_type(16))) _Float16 v16h;
typedef __attribute__((ext_vector_type(8)))  float    v8f;

#define HW      65536     // H*W = 256*256
#define NBATCH  32
#define NCHUNK  32        // blocks per batch (gridDim.x)
#define PXTHR   8         // pixels per thread (2 x float4)
#define NPART   7         // partial slots: C0,C1,O0,O1,B0,B1,P

// ---- fast hardware transcendentals (guarded; fallbacks keep compile safe) ----
__device__ __forceinline__ float fexp(float x) {          // e^x
#if __has_builtin(__builtin_amdgcn_exp2f)
    return __builtin_amdgcn_exp2f(x * 1.44269504088896340736f);
#else
    return expf(x);
#endif
}
__device__ __forceinline__ float frcp(float x) {          // 1/x (~1 ulp)
#if __has_builtin(__builtin_amdgcn_rcpf)
    return __builtin_amdgcn_rcpf(x);
#else
    return 1.0f / x;
#endif
}
__device__ __forceinline__ float flog(float x) {          // ln(x)
#if __has_builtin(__builtin_amdgcn_logf)
    return __builtin_amdgcn_logf(x) * 0.69314718055994530942f;
#else
    return logf(x);
#endif
}
__device__ __forceinline__ float ftanh(float x) {         // tanh(x)
#if __has_builtin(__builtin_amdgcn_tanhf)
    return __builtin_amdgcn_tanhf(x);                     // gfx1250 v_tanh_f32
#else
    // tanh(x) = 1 - 2/(e^{2x}+1); exact at +/-inf limits
    return 1.0f - 2.0f * frcp(fexp(2.0f * x) + 1.0f);
#endif
}
__device__ __forceinline__ float fsigmoid(float x) {
    return frcp(1.0f + fexp(-x));
}

__device__ __forceinline__ float f4get(float4 v, int i) {
    return (i == 0) ? v.x : (i == 1) ? v.y : (i == 2) ? v.z : v.w;
}
__device__ __forceinline__ float4 ld4(const float* p) {
    return *reinterpret_cast<const float4*>(p);
}

__device__ __forceinline__ float wave_sum32(float v) {
#pragma unroll
    for (int m = 16; m > 0; m >>= 1) v += __shfl_xor(v, m, 32);
    return v;
}

// ---------------------------------------------------------------------------
// Kernel A: fused per-pixel reduction over both stacks.
// grid = (NCHUNK, NBATCH), block = 256 (8 wave32). Block covers 2048 pixels
// of batch b; thread t handles pixels q = blockBase + h*1024 + t*4 + (0..3),
// so every global_load_b128 is a fully contiguous 512B wave transaction.
// Accumulates (per stack): corner CE, masked smooth-L1, bin |err|*pos,
// plus pos-count via V_WMMA_F32_16X16X32_F16 (exact: 0/1 in f16, f32 acc).
// ---------------------------------------------------------------------------
__global__ __launch_bounds__(256) void mt_reduce_kernel(
    const float* __restrict__ out0, const float* __restrict__ out1,
    const float* __restrict__ t_corner, const float* __restrict__ t_co,
    const float* __restrict__ t_bin, float* __restrict__ partials)
{
    const int b = blockIdx.y;
    const float* op0  = out0 + (size_t)b * 7 * HW;
    const float* op1  = out1 + (size_t)b * 7 * HW;
    const float* posb = t_corner + (size_t)b * HW;
    const float* cob  = t_co  + (size_t)b * 2 * HW;
    const float* tbb  = t_bin + (size_t)b * 2 * HW;

    const int t         = threadIdx.x;
    const int blockBase = blockIdx.x * 2048;

    float accC[2] = {0.f, 0.f};   // corner cross-entropy
    float accO[2] = {0.f, 0.f};   // masked smooth-L1 (d0+d1 folded)
    float accB[2] = {0.f, 0.f};   // bin offset |err| * pos (d0+d1 folded)
    v16h aPos = {};               // 8 pos values + 8 zeros (zeros add nothing)

#pragma unroll
    for (int h = 0; h < 2; ++h) {
        const int q = blockBase + h * 1024 + t * 4;

        // speculative prefetch of the next block's lines (global_prefetch_b8)
        __builtin_prefetch(op0 + 1 * HW + q + 2048, 0, 0);
        __builtin_prefetch(op1 + 1 * HW + q + 2048, 0, 0);

        const float4 pos4 = ld4(posb + q);
        const float4 tg0v = ld4(cob + q);
        const float4 tg1v = ld4(cob + HW + q);
        const float4 tb0v = ld4(tbb + q);
        const float4 tb1v = ld4(tbb + HW + q);

        float4 c1v[2], c2v[2], o3v[2], o4v[2], o5v[2], o6v[2];
#pragma unroll
        for (int s = 0; s < 2; ++s) {
            const float* op = s ? op1 : op0;
            c1v[s] = ld4(op + 1 * HW + q);
            c2v[s] = ld4(op + 2 * HW + q);
            o3v[s] = ld4(op + 3 * HW + q);
            o4v[s] = ld4(op + 4 * HW + q);
            o5v[s] = ld4(op + 5 * HW + q);
            o6v[s] = ld4(op + 6 * HW + q);
        }

#pragma unroll
        for (int j = 0; j < 4; ++j) {
            const float pos = f4get(pos4, j);
            aPos[h * 4 + j] = (_Float16)pos;     // exact: pos is 0.0 or 1.0
            const float tg0 = f4get(tg0v, j);
            const float tg1 = f4get(tg1v, j);
            const float tb0 = f4get(tb0v, j);
            const float tb1 = f4get(tb1v, j);

#pragma unroll
            for (int s = 0; s < 2; ++s) {
                // corner: -log_softmax over channels (1,2)
                const float c1 = f4get(c1v[s], j);
                const float c2 = f4get(c2v[s], j);
                const float mx = fmaxf(c1, c2);
                const float lse = mx + flog(1.0f + fexp(-fabsf(c1 - c2)));
                accC[s] += pos * (lse - c2) * 10000.f
                         + (1.f - pos) * (lse - c1);

                // corner_offset: tanh*256, masked smooth-L1, dims 0,1
                {
                    const float df = ftanh(f4get(o3v[s], j)) * 256.f - tg0;
                    const float ad = fabsf(df);
                    const float sl = (ad < 1.f) ? 0.5f * df * df : ad - 0.5f;
                    accO[s] += ((tg0 != 0.f) ? 10000.f : 1.f) * sl;
                }
                {
                    const float df = ftanh(f4get(o4v[s], j)) * 256.f - tg1;
                    const float ad = fabsf(df);
                    const float sl = (ad < 1.f) ? 0.5f * df * df : ad - 0.5f;
                    accO[s] += ((tg1 != 0.f) ? 10000.f : 1.f) * sl;
                }

                // corner_bin: |sigmoid - 0.5 - tb| * pos, dims 0,1
                accB[s] += fabsf(fsigmoid(f4get(o5v[s], j)) - 0.5f - tb0) * pos;
                accB[s] += fabsf(fsigmoid(f4get(o6v[s], j)) - 0.5f - tb1) * pos;
            }
        }
    }

    // pos count on the matrix pipe: D = A(pos,f16) x ones(f16) + 0, f32 acc.
    // B all-ones => every D column = total; sum of all 256 D entries = 16*total.
    v16h ones;
#pragma unroll
    for (int j = 0; j < 16; ++j) ones[j] = (_Float16)1.0f;
    v8f c = {};
    c = __builtin_amdgcn_wmma_f32_16x16x32_f16(
        /*neg_a=*/false, aPos, /*neg_b=*/false, ones,
        /*c_mod=*/(short)0, c, /*reuse_a=*/false, /*reuse_b=*/false);
    float vs = 0.f;
#pragma unroll
    for (int r = 0; r < 8; ++r) vs += c[r];

    float w0 = wave_sum32(accC[0]);
    float w1 = wave_sum32(accC[1]);
    float w2 = wave_sum32(accO[0]);
    float w3 = wave_sum32(accO[1]);
    float w4 = wave_sum32(accB[0]);
    float w5 = wave_sum32(accB[1]);
    float w6 = wave_sum32(vs) * (1.0f / 16.0f);   // exact integer count

    __shared__ float red[8][NPART];
    const int wave = t >> 5;
    const int lane = t & 31;
    if (lane == 0) {
        red[wave][0] = w0; red[wave][1] = w1; red[wave][2] = w2;
        red[wave][3] = w3; red[wave][4] = w4; red[wave][5] = w5;
        red[wave][6] = w6;
    }
    __syncthreads();
    if (t == 0) {
        float s0 = 0, s1 = 0, s2 = 0, s3 = 0, s4 = 0, s5 = 0, s6 = 0;
#pragma unroll
        for (int w = 0; w < 8; ++w) {
            s0 += red[w][0]; s1 += red[w][1]; s2 += red[w][2];
            s3 += red[w][3]; s4 += red[w][4]; s5 += red[w][5];
            s6 += red[w][6];
        }
        float* dst = partials + ((size_t)b * NCHUNK + blockIdx.x) * NPART;
        dst[0] = s0; dst[1] = s1; dst[2] = s2; dst[3] = s3;
        dst[4] = s4; dst[5] = s5; dst[6] = s6;
    }
}

// ---------------------------------------------------------------------------
// Kernel B: grid = 2 (stack), block = 32 (one wave, thread = batch).
// Folds chunk partials, does triplet-center gathers, writes all 194 outputs.
// ---------------------------------------------------------------------------
__global__ __launch_bounds__(32) void mt_final_kernel(
    const float* __restrict__ out0, const float* __restrict__ out1,
    const float* __restrict__ partials,
    const int* __restrict__ anchors, const int* __restrict__ positives,
    const int* __restrict__ neg_zero, const int* __restrict__ neg_other,
    float* __restrict__ out)
{
    const int s = blockIdx.x;
    const int b = threadIdx.x;   // 0..31

    float sc = 0.f, so = 0.f, sb = 0.f, sp = 0.f;
    const float* pp = partials + (size_t)b * NCHUNK * NPART;
    for (int c = 0; c < NCHUNK; ++c) {
        const float* q = pp + c * NPART;
        sc += q[0 + s];
        so += q[2 + s];
        sb += q[4 + s];
        sp += q[6];
    }

    const float inv = 1.0f / (float)HW;
    out[s * 97 + 1 + b]  = sc * inv;                 // l_corner * 1.0
    out[s * 97 + 33 + b] = so * inv * 0.25f;         // l_co * 0.25
    float w = sp * inv;
    if (w == 0.f) w = 1.f;
    out[s * 97 + 65 + b] = (sb * inv) / w;           // l_bo * 1.0

    // triplet center loss on channel 0
    const float* ctr = (s ? out1 : out0) + (size_t)b * 7 * HW;
    const int* an = anchors   + b * 9;
    const int* po = positives + b * 9;
    const int* nz = neg_zero  + b * 9;
    const int* no = neg_other + b * 8;
    float cs = 0.f;
#pragma unroll
    for (int i = 0; i < 9; ++i) {
        const float vA  = ctr[an[i]];
        const float vP  = ctr[po[i]];
        const float vN1 = ctr[nz[i]];
        const float Dap = (vA - vP) * (vA - vP);
        const float h1  = fmaxf(Dap - (vA - vN1) * (vA - vN1) + 10.f, 0.f);
        float per;
        if (i == 0) {
            per = h1;
        } else {
            const float vN2 = ctr[no[i - 1]];
            const float h2  = fmaxf(Dap - (vA - vN2) * (vA - vN2) + 10.f, 0.f);
            per = 0.5f * (h1 + h2);
        }
        cs += per;
    }
    cs *= (1.0f / 9.0f);           // / (K+1)
    cs = wave_sum32(cs);
    if (b == 0) out[s * 97] = cs * (1.0f / 32.0f);   // mean over B, * 1.0
}

// ---------------------------------------------------------------------------
extern "C" void kernel_launch(void* const* d_in, const int* in_sizes, int n_in,
                              void* d_out, int out_size, void* d_ws, size_t ws_size,
                              hipStream_t stream) {
    (void)in_sizes; (void)n_in; (void)out_size; (void)ws_size;
    const float* out0      = (const float*)d_in[0];
    const float* out1      = (const float*)d_in[1];
    const float* t_corner  = (const float*)d_in[2];
    const float* t_co      = (const float*)d_in[3];
    const float* t_bin     = (const float*)d_in[4];
    const int*   anchors   = (const int*)d_in[5];
    const int*   positives = (const int*)d_in[6];
    const int*   neg_zero  = (const int*)d_in[7];
    const int*   neg_other = (const int*)d_in[8];
    float*       out       = (float*)d_out;
    float*       partials  = (float*)d_ws;   // NBATCH*NCHUNK*NPART floats = 28 KB

    dim3 gridA(NCHUNK, NBATCH);
    mt_reduce_kernel<<<gridA, 256, 0, stream>>>(out0, out1, t_corner, t_co,
                                                t_bin, partials);
    mt_final_kernel<<<2, 32, 0, stream>>>(out0, out1, partials, anchors,
                                          positives, neg_zero, neg_other, out);
}
